// IGASlotPoolingV1_67834713473719
// MI455X (gfx1250) — compile-verified, matrix operand order checked by reading, not run on hardware
//
#include <hip/hip_runtime.h>
#include <math.h>

// ---------------- problem constants ----------------
constexpr int cB = 4, cN = 2048, cC = 256, cKM = 1024, cKU = 171, cH = 4;
constexpr float cEPS = 1e-8f, cSIG_FLOOR = 0.03f, cINIT_SIG_MIX = 0.1f;
constexpr float cINIT_OFF = 0.5f, cJIT = 1e-6f;
constexpr float cW_KL = 0.05f, cW_ENT = 0.0f, cW_OCC = 0.5f, cW_COL = 0.1f;

// ---------------- output layout (floats) ----------------
constexpr long long OFF_A     = 0;
constexpr long long OFF_SLOTS = OFF_A     + (long long)cB*cN*cKU;
constexpr long long OFF_MUK   = OFF_SLOTS + (long long)cB*cKU*cC;
constexpr long long OFF_SIGK  = OFF_MUK   + (long long)cB*cKU*3;
constexpr long long OFF_IDX   = OFF_SIGK  + (long long)cB*cKU*9;
constexpr long long OFF_KL    = OFF_IDX   + (long long)cB*cKU;
constexpr long long OFF_ENT   = OFF_KL  + 1;
constexpr long long OFF_OMSE  = OFF_ENT + 1;
constexpr long long OFF_COL   = OFF_OMSE + 1;
constexpr long long OFF_TOT   = OFF_COL + 1;

typedef __attribute__((ext_vector_type(16))) _Float16 v16h;
typedef __attribute__((ext_vector_type(8)))  float    v8f;
typedef __attribute__((ext_vector_type(4)))  float    v4f;

// Let InferAddressSpaces promote flat->global accesses.
__device__ __forceinline__ const float* gptr(const float* p) {
  __builtin_assume(!__builtin_amdgcn_is_shared((const void*)p));
  __builtin_assume(!__builtin_amdgcn_is_private((const void*)p));
  return p;
}
__device__ __forceinline__ float* gptr(float* p) {
  __builtin_assume(!__builtin_amdgcn_is_shared((const void*)p));
  __builtin_assume(!__builtin_amdgcn_is_private((const void*)p));
  return p;
}

// ---------------- block reduction helper (256 threads) ----------------
__device__ __forceinline__ float bsum(float v, float* sh) {
  int t = threadIdx.x;
  sh[t] = v; __syncthreads();
#pragma unroll
  for (int s = 128; s > 0; s >>= 1) { if (t < s) sh[t] += sh[t + s]; __syncthreads(); }
  float r = sh[0]; __syncthreads();
  return r;
}

// ---------------- fast WMMA GEMM: contiguous K (sAk==sBk==1) ----------------
// One wave computes a 64x16 macro-tile (4 M-tiles), reusing the B fragment 4x.
// C[b][m][n] = sum_k A[b][m][k] * B[b][n][k] (+ bias[n])
__global__ __launch_bounds__(32) void wmma_gemm_fast(
    const float* __restrict__ A, const float* __restrict__ B,
    const float* __restrict__ bias, float* __restrict__ Cout,
    int M, int Nn, int K,
    long long sAm, long long sBn, long long sCm,
    long long bsA, long long bsB, long long bsC)
{
  int batch = blockIdx.y;
  int tilesN = (Nn + 15) >> 4;
  int tm = blockIdx.x / tilesN, tn = blockIdx.x % tilesN;
  int lane = threadIdx.x & 31;
  int half = lane >> 4;      // lane group
  int lid  = lane & 15;

  int bn = tn * 16 + lid;    // B row (output column) held by this lane
  float bS = (bn < Nn) ? 1.0f : 0.0f;
  const float* Bp = gptr(B) + batch * bsB + (long long)min(bn, Nn - 1) * sBn + half * 8;

  int am0 = tm * 64 + 0  * 16 + lid;
  int am1 = tm * 64 + 1  * 16 + lid;
  int am2 = tm * 64 + 2  * 16 + lid;
  int am3 = tm * 64 + 3  * 16 + lid;
  float aS0 = (am0 < M) ? 1.0f : 0.0f;
  float aS1 = (am1 < M) ? 1.0f : 0.0f;
  float aS2 = (am2 < M) ? 1.0f : 0.0f;
  float aS3 = (am3 < M) ? 1.0f : 0.0f;
  const float* Ag = gptr(A) + batch * bsA + half * 8;
  const float* Ap0 = Ag + (long long)min(am0, M - 1) * sAm;
  const float* Ap1 = Ag + (long long)min(am1, M - 1) * sAm;
  const float* Ap2 = Ag + (long long)min(am2, M - 1) * sAm;
  const float* Ap3 = Ag + (long long)min(am3, M - 1) * sAm;

  v8f z = {0.f,0.f,0.f,0.f,0.f,0.f,0.f,0.f};
  v8f acc0 = z, acc1 = z, acc2 = z, acc3 = z;

#define LOAD_FRAG(fr, P, S)                                                   \
  {                                                                           \
    v4f x0 = *(const v4f*)((P) + 0);                                          \
    v4f x1 = *(const v4f*)((P) + 4);                                          \
    v4f x2 = *(const v4f*)((P) + 16);                                         \
    v4f x3 = *(const v4f*)((P) + 20);                                         \
    __builtin_prefetch((const void*)((P) + 32), 0, 3);                        \
    _Pragma("unroll")                                                         \
    for (int j = 0; j < 4; ++j) {                                             \
      fr[j]      = (_Float16)(x0[j] * (S));                                   \
      fr[j + 4]  = (_Float16)(x1[j] * (S));                                   \
      fr[j + 8]  = (_Float16)(x2[j] * (S));                                   \
      fr[j + 12] = (_Float16)(x3[j] * (S));                                   \
    }                                                                         \
  }

  for (int k0 = 0; k0 < K; k0 += 32) {
    v16h bf, af;
    LOAD_FRAG(bf, Bp, bS);
    LOAD_FRAG(af, Ap0, aS0);
    acc0 = __builtin_amdgcn_wmma_f32_16x16x32_f16(false, af, false, bf,
                                                  (short)0, acc0, false, false);
    LOAD_FRAG(af, Ap1, aS1);
    acc1 = __builtin_amdgcn_wmma_f32_16x16x32_f16(false, af, false, bf,
                                                  (short)0, acc1, false, false);
    LOAD_FRAG(af, Ap2, aS2);
    acc2 = __builtin_amdgcn_wmma_f32_16x16x32_f16(false, af, false, bf,
                                                  (short)0, acc2, false, false);
    LOAD_FRAG(af, Ap3, aS3);
    acc3 = __builtin_amdgcn_wmma_f32_16x16x32_f16(false, af, false, bf,
                                                  (short)0, acc3, false, false);
    Ap0 += 32; Ap1 += 32; Ap2 += 32; Ap3 += 32; Bp += 32;
  }
#undef LOAD_FRAG

  int cn = tn * 16 + lid;
  float* Cb = gptr(Cout) + batch * bsC;
  if (cn < Nn) {
    float bval = bias ? gptr(bias)[cn] : 0.0f;
    v8f accs[4] = {acc0, acc1, acc2, acc3};
#pragma unroll
    for (int mi = 0; mi < 4; ++mi) {
#pragma unroll
      for (int r = 0; r < 8; ++r) {
        int cm = tm * 64 + mi * 16 + half * 8 + r;
        if (cm < M)
          Cb[(long long)cm * sCm + cn] = accs[mi][r] + bval;
      }
    }
  }
}

// ---------------- strided WMMA GEMM (transposed operands) ----------------
__global__ __launch_bounds__(32) void wmma_gemm_strided(
    const float* __restrict__ A, const float* __restrict__ B,
    float* __restrict__ Cout,
    int M, int Nn, int K,
    long long sAm, long long sAk, long long sBn, long long sBk,
    long long sCm, long long sCn,
    long long bsA, long long bsB, long long bsC)
{
  int batch = blockIdx.y;
  int tilesN = (Nn + 15) >> 4;
  int tm = blockIdx.x / tilesN, tn = blockIdx.x % tilesN;
  int lane = threadIdx.x & 31;
  int half = lane >> 4;
  int lid  = lane & 15;

  int am = tm * 16 + lid;
  int bn = tn * 16 + lid;
  float aS = (am < M) ? 1.0f : 0.0f;
  float bS = (bn < Nn) ? 1.0f : 0.0f;

  const float* Ap = gptr(A) + batch * bsA + (long long)min(am, M - 1) * sAm + (long long)(half * 8) * sAk;
  const float* Bp = gptr(B) + batch * bsB + (long long)min(bn, Nn - 1) * sBn + (long long)(half * 8) * sBk;

  v8f acc = {0.f,0.f,0.f,0.f,0.f,0.f,0.f,0.f};
  for (int k0 = 0; k0 < K; k0 += 32) {
    v16h af, bf;
#pragma unroll
    for (int h = 0; h < 16; ++h) {
      int kr = (h < 8) ? h : (h + 8);  // two runs: [0..7], [16..23]
      af[h] = (_Float16)(Ap[(long long)kr * sAk] * aS);
      bf[h] = (_Float16)(Bp[(long long)kr * sBk] * bS);
    }
    acc = __builtin_amdgcn_wmma_f32_16x16x32_f16(false, af, false, bf,
                                                 (short)0, acc, false, false);
    Ap += 32 * sAk; Bp += 32 * sBk;
  }

  int cn = tn * 16 + lid;
  float* Cb = gptr(Cout) + batch * bsC;
  if (cn < Nn) {
#pragma unroll
    for (int r = 0; r < 8; ++r) {
      int cm = tm * 16 + half * 8 + r;
      if (cm < M)
        Cb[(long long)cm * sCm + (long long)cn * sCn] = acc[r];
    }
  }
}

// ---------------- per-batch statistics ----------------
__global__ __launch_bounds__(256) void k_stats(
    const float* __restrict__ s, const float* __restrict__ mu,
    const float* __restrict__ Sg, const float* __restrict__ mask,
    float* s_global, float* meanO, float* stdO, float* sigavgO)
{
  __shared__ float sh[256];
  int b = blockIdx.x, t = threadIdx.x;
  const float* sb  = gptr(s)  + (long long)b * cN * cC;
  const float* mub = gptr(mu) + (long long)b * cN * 3;
  const float* Sb  = gptr(Sg) + (long long)b * cN * 9;
  const float* mb  = gptr(mask) + (long long)b * cN;

  float p = 0.f;
  for (int n = t; n < cN; n += 256) p += mb[n];
  float cnt = fmaxf(bsum(p, sh), cEPS);

  { // s_global: thread t owns channel t (cC == 256)
    float a = 0.f;
    for (int n = 0; n < cN; ++n) a += sb[(long long)n * cC + t] * mb[n];
    s_global[b * cC + t] = a / cnt;
  }

  float m3[3];
  for (int i = 0; i < 3; ++i) {
    p = 0.f;
    for (int n = t; n < cN; n += 256) p += mub[n * 3 + i] * mb[n];
    m3[i] = bsum(p, sh) / cnt;
  }
  if (t == 0) for (int i = 0; i < 3; ++i) meanO[b * 3 + i] = m3[i];

  float dg[3];
  for (int i = 0; i < 3; ++i) {
    p = 0.f;
    for (int n = t; n < cN; n += 256) {
      float d = mub[n * 3 + i] - m3[i];
      p += d * d * mb[n];
    }
    dg[i] = bsum(p, sh) / cnt;
  }
  if (t == 0)
    for (int i = 0; i < 3; ++i)
      stdO[b * 3 + i] = fmaxf(sqrtf(fmaxf(dg[i], cEPS)), cSIG_FLOOR);

  for (int e = 0; e < 9; ++e) {
    p = 0.f;
    for (int n = t; n < cN; n += 256) p += Sb[n * 9 + e] * mb[n];
    float v = bsum(p, sh) / cnt;
    if (t == 0) sigavgO[b * 9 + e] = v;
  }
}

// ---------------- gating scores ----------------
__global__ __launch_bounds__(256) void k_gating(
    const float* __restrict__ s_global, const float* __restrict__ gW,
    const float* __restrict__ pool, float* scores)
{
  __shared__ float sh[256];
  __shared__ float qbuf[cC];
  __shared__ float qn;
  int b = blockIdx.x, t = threadIdx.x;
  const float* sg = gptr(s_global) + b * cC;
  const float* wr = gptr(gW) + (long long)t * cC;
  float a = 0.f;
  for (int j = 0; j < cC; ++j) a += sg[j] * wr[j];
  qbuf[t] = a;
  float nrm = bsum(a * a, sh);
  if (t == 0) qn = fmaxf(sqrtf(nrm), 1e-12f);
  __syncthreads();
  for (int r = t; r < cKM; r += 256) {
    const float* pr = gptr(pool) + (long long)r * cC;
    float d = 0.f, n2 = 0.f;
    for (int j = 0; j < cC; ++j) { float pv = pr[j]; d += qbuf[j] * pv; n2 += pv * pv; }
    scores[b * cKM + r] = d / (qn * fmaxf(sqrtf(n2), 1e-12f));
  }
}

// ---------------- top-K selection + gather ----------------
__global__ __launch_bounds__(256) void k_topk(
    const float* __restrict__ scores, const float* __restrict__ pool,
    const float* __restrict__ gpool,
    int* idxI, float* idxF, float* slots, float* geo_off)
{
  __shared__ float sv[cKM];
  __shared__ float rv[256];
  __shared__ int   ri[256];
  int b = blockIdx.x, t = threadIdx.x;
  for (int r = t; r < cKM; r += 256) sv[r] = scores[b * cKM + r];
  __syncthreads();
  for (int k = 0; k < cKU; ++k) {
    float best = -3e38f; int bi = cKM;
    for (int r = t; r < cKM; r += 256) {
      float v = sv[r];
      if (v > best || (v == best && r < bi)) { best = v; bi = r; }
    }
    rv[t] = best; ri[t] = bi; __syncthreads();
    for (int s2 = 128; s2 > 0; s2 >>= 1) {
      if (t < s2) {
        if (rv[t + s2] > rv[t] || (rv[t + s2] == rv[t] && ri[t + s2] < ri[t])) {
          rv[t] = rv[t + s2]; ri[t] = ri[t + s2];
        }
      }
      __syncthreads();
    }
    if (t == 0) {
      idxI[b * cKU + k] = ri[0];
      idxF[b * cKU + k] = (float)ri[0];
      sv[ri[0]] = -3e38f;
    }
    __syncthreads();
  }
  for (int e = t; e < cKU * cC; e += 256) {
    int k = e / cC, c = e % cC;
    slots[((long long)b * cKU + k) * cC + c] = pool[(long long)idxI[b * cKU + k] * cC + c];
  }
  for (int e = t; e < cKU * 3; e += 256) {
    int k = e / 3, c = e % 3;
    geo_off[((long long)b * cKU + k) * 3 + c] = gpool[(long long)idxI[b * cKU + k] * 3 + c];
  }
}

// ---------------- cluster init ----------------
__global__ __launch_bounds__(256) void k_init(
    const float* meanw, const float* stdw, const float* sigavg,
    const float* geo_off, float* mu_k, float* sig_k)
{
  int b = blockIdx.x, t = threadIdx.x;
  for (int e = t; e < cKU * 3; e += 256) {
    int k = e / 3, i = e % 3;
    mu_k[((long long)b * cKU + k) * 3 + i] =
        meanw[b * 3 + i] + geo_off[((long long)b * cKU + k) * 3 + i] * (cINIT_OFF * stdw[b * 3 + i]);
  }
  for (int e = t; e < cKU * 9; e += 256) {
    int k = e / 9, ij = e % 9, i = ij / 3, j = ij % 3;
    float v = sigavg[b * 9 + ij] * cINIT_SIG_MIX +
              ((i == j) ? (cSIG_FLOOR * cSIG_FLOOR + cJIT) : 0.0f);
    sig_k[((long long)b * cKU + k) * 9 + ij] = v;
  }
}

// ---------------- geo term + masked softmax (one wave32 per (b,n) row) ----------------
__global__ __launch_bounds__(32) void k_geo_softmax(
    const float* __restrict__ lsem, const float* __restrict__ mu,
    const float* __restrict__ Sg, const float* __restrict__ mask,
    const float* __restrict__ mu_k, const float* __restrict__ sig_k,
    const float* __restrict__ gsr, const float* __restrict__ gbias,
    const float* __restrict__ wgeo, float* __restrict__ A_out)
{
  int row = blockIdx.x;
  int b = row / cN, n = row % cN;
  int lane = threadIdx.x;

  float sA = 0.f, sB = 0.f;
#pragma unroll
  for (int h = 0; h < cH; ++h) {
    float x = gsr[h];
    sA += logf(1.0f + expf(x));   // softplus
    sB += gbias[h];
  }
  float wg = wgeo[0];
  float m  = mask[b * cN + n];

  const float* mun = gptr(mu) + ((long long)b * cN + n) * 3;
  const float* Sn  = gptr(Sg) + ((long long)b * cN + n) * 9;
  float u0 = mun[0], u1 = mun[1], u2 = mun[2];
  float n00 = Sn[0], n11 = Sn[4], n22 = Sn[8];
  float n01 = 0.5f * (Sn[1] + Sn[3]);
  float n02 = 0.5f * (Sn[2] + Sn[6]);
  float n12 = 0.5f * (Sn[5] + Sn[7]);

  const float invSq = 0.0625f; // 1/sqrt(256)
  float lv[6];
  int nk = 0;
  float lmax = -3e38f;
  for (int k = lane; k < cKU; k += 32) {
    const float* mk = gptr(mu_k) + ((long long)b * cKU + k) * 3;
    const float* Sk = gptr(sig_k) + ((long long)b * cKU + k) * 9;
    float d0 = u0 - mk[0], d1 = u1 - mk[1], d2 = u2 - mk[2];
    float s00 = n00 + Sk[0], s11 = n11 + Sk[4], s22 = n22 + Sk[8];
    float s01 = n01 + 0.5f * (Sk[1] + Sk[3]);
    float s02 = n02 + 0.5f * (Sk[2] + Sk[6]);
    float s12 = n12 + 0.5f * (Sk[5] + Sk[7]);
    float i00 = s11 * s22 - s12 * s12;
    float i01 = s02 * s12 - s01 * s22;
    float i02 = s01 * s12 - s02 * s11;
    float i11 = s00 * s22 - s02 * s02;
    float i12 = s01 * s02 - s00 * s12;
    float i22 = s00 * s11 - s01 * s01;
    float det = s00 * i00 + s01 * i01 + s02 * i02;
    float detg = fmaxf(det, 1e-30f);
    float quad = d0 * (i00 * d0 + i01 * d1 + i02 * d2) +
                 d1 * (i01 * d0 + i11 * d1 + i12 * d2) +
                 d2 * (i02 * d0 + i12 * d1 + i22 * d2);
    float maha = quad / detg;
    float G = -0.5f * (maha + logf(detg));
    float geo = sA * G + sB;
    float L = lsem[((long long)b * cN + n) * cKU + k] * invSq + wg * geo;
    if (m < 0.5f) L = -1e9f;
    lv[nk++] = L;
    lmax = fmaxf(lmax, L);
  }
#pragma unroll
  for (int o = 16; o > 0; o >>= 1) lmax = fmaxf(lmax, __shfl_xor(lmax, o, 32));
  float lsum = 0.f;
  for (int i = 0; i < nk; ++i) { lv[i] = expf(lv[i] - lmax); lsum += lv[i]; }
#pragma unroll
  for (int o = 16; o > 0; o >>= 1) lsum += __shfl_xor(lsum, o, 32);
  float inv = 1.0f / lsum;
  int i2 = 0;
  for (int k = lane; k < cKU; k += 32)
    A_out[((long long)b * cN + n) * cKU + k] = lv[i2++] * inv * m;
}

// ---------------- cluster moment update ----------------
__global__ __launch_bounds__(256) void k_update(
    const float* __restrict__ A, const float* __restrict__ mu,
    const float* __restrict__ Sg, float* mu_k, float* sig_k, float* occ)
{
  __shared__ float sh[256];
  int b = blockIdx.x / cKU, k = blockIdx.x % cKU, t = threadIdx.x;
  const float* Ab = gptr(A) + (long long)b * cN * cKU + k;
  const float* mub = gptr(mu) + (long long)b * cN * 3;
  const float* Sb  = gptr(Sg) + (long long)b * cN * 9;

  float po = 0.f, p0 = 0.f, p1 = 0.f, p2 = 0.f;
  for (int n = t; n < cN; n += 256) {
    float a = Ab[(long long)n * cKU];
    po += a;
    p0 += a * mub[n * 3 + 0];
    p1 += a * mub[n * 3 + 1];
    p2 += a * mub[n * 3 + 2];
  }
  float occv = fmaxf(bsum(po, sh), cEPS);
  float mk0 = bsum(p0, sh) / occv;
  float mk1 = bsum(p1, sh) / occv;
  float mk2 = bsum(p2, sh) / occv;
  if (t == 0) {
    occ[b * cKU + k] = occv;
    mu_k[((long long)b * cKU + k) * 3 + 0] = mk0;
    mu_k[((long long)b * cKU + k) * 3 + 1] = mk1;
    mu_k[((long long)b * cKU + k) * 3 + 2] = mk2;
  }

  float q00 = 0.f, q01 = 0.f, q02 = 0.f, q11 = 0.f, q12 = 0.f, q22 = 0.f;
  for (int n = t; n < cN; n += 256) {
    float a = Ab[(long long)n * cKU];
    float d0 = mub[n * 3 + 0] - mk0;
    float d1 = mub[n * 3 + 1] - mk1;
    float d2 = mub[n * 3 + 2] - mk2;
    const float* S = Sb + (long long)n * 9;
    q00 += a * (S[0] + d0 * d0);
    q11 += a * (S[4] + d1 * d1);
    q22 += a * (S[8] + d2 * d2);
    q01 += a * (0.5f * (S[1] + S[3]) + d0 * d1);
    q02 += a * (0.5f * (S[2] + S[6]) + d0 * d2);
    q12 += a * (0.5f * (S[5] + S[7]) + d1 * d2);
  }
  float r00 = bsum(q00, sh) / occv;
  float r01 = bsum(q01, sh) / occv;
  float r02 = bsum(q02, sh) / occv;
  float r11 = bsum(q11, sh) / occv;
  float r12 = bsum(q12, sh) / occv;
  float r22 = bsum(q22, sh) / occv;
  if (t == 0) {
    float f00 = fmaxf(r00 + cJIT, cSIG_FLOOR * cSIG_FLOOR) + cJIT;
    float f11 = fmaxf(r11 + cJIT, cSIG_FLOOR * cSIG_FLOOR) + cJIT;
    float f22 = fmaxf(r22 + cJIT, cSIG_FLOOR * cSIG_FLOOR) + cJIT;
    float* S = sig_k + ((long long)b * cKU + k) * 9;
    S[0] = f00; S[4] = f11; S[8] = f22;
    S[1] = S[3] = r01; S[2] = S[6] = r02; S[5] = S[7] = r12;
  }
}

// ---------------- elementwise kernels ----------------
__global__ void k_zero(float* p, long long n) {
  for (long long i = (long long)blockIdx.x * blockDim.x + threadIdx.x; i < n;
       i += (long long)gridDim.x * blockDim.x) p[i] = 0.f;
}
__global__ void k_divocc(float* slot_in, const float* occ) {
  long long tot = (long long)cB * cKU * cC;
  for (long long i = (long long)blockIdx.x * blockDim.x + threadIdx.x; i < tot;
       i += (long long)gridDim.x * blockDim.x) slot_in[i] /= occ[i / cC];
}
__global__ void k_gru(const float* __restrict__ gi, const float* __restrict__ gh,
                      float* __restrict__ slots) {
  long long tot = (long long)cB * cKU * cC;
  for (long long i = (long long)blockIdx.x * blockDim.x + threadIdx.x; i < tot;
       i += (long long)gridDim.x * blockDim.x) {
    long long row = i / cC; int c = (int)(i % cC);
    const float* gir = gptr(gi) + row * 3 * cC;
    const float* ghr = gptr(gh) + row * 3 * cC;
    float r = 1.0f / (1.0f + expf(-(gir[c] + ghr[c])));
    float z = 1.0f / (1.0f + expf(-(gir[cC + c] + ghr[cC + c])));
    float nn = tanhf(gir[2 * cC + c] + r * ghr[2 * cC + c]);
    float h = slots[i];
    slots[i] = (1.0f - z) * nn + z * h;
  }
}
__global__ __launch_bounds__(256) void k_ln(const float* __restrict__ slots,
                                            const float* __restrict__ g,
                                            const float* __restrict__ bb,
                                            float* __restrict__ xm) {
  __shared__ float sh[256];
  long long row = blockIdx.x; int t = threadIdx.x;
  float x = slots[row * cC + t];
  float mean = bsum(x, sh) / cC;
  float d = x - mean;
  float var = bsum(d * d, sh) / cC;
  xm[row * cC + t] = d * rsqrtf(var + 1e-5f) * g[t] + bb[t];
}
__global__ void k_gelu(float* p) {
  long long tot = (long long)cB * cKU * 4 * cC;
  for (long long i = (long long)blockIdx.x * blockDim.x + threadIdx.x; i < tot;
       i += (long long)gridDim.x * blockDim.x) {
    float x = p[i];
    p[i] = 0.5f * x * (1.0f + erff(x * 0.70710678118f));
  }
}
__global__ void k_add(float* slots, const float* mo) {
  long long tot = (long long)cB * cKU * cC;
  for (long long i = (long long)blockIdx.x * blockDim.x + threadIdx.x; i < tot;
       i += (long long)gridDim.x * blockDim.x) slots[i] += mo[i];
}

// ---------------- losses ----------------
__global__ void k_hist(const int* idxI, float* hist) {
  int i = blockIdx.x * blockDim.x + threadIdx.x;
  if (i < cB * cKU) atomicAdd(&hist[idxI[i]], 1.0f);
}
__global__ __launch_bounds__(256) void k_usage(const float* hist, float* out) {
  __shared__ float sh[256];
  int t = threadIdx.x;
  float p = 0.f;
  for (int i = t; i < cKM; i += 256) p += hist[i];
  float hs = fmaxf(bsum(p, sh), 1.0f);
  float kl = 0.f, en = 0.f;
  const float lg = logf(1.0f / (float)cKM);
  for (int i = t; i < cKM; i += 256) {
    float pc = fmaxf(hist[i] / hs, 1e-8f);
    float lp = logf(pc);
    kl += pc * (lp - lg);
    en += pc * lp;
  }
  kl = bsum(kl, sh);
  en = bsum(en, sh);
  if (t == 0) { out[OFF_KL] = kl; out[OFF_ENT] = en; }
}
__global__ __launch_bounds__(256) void k_colstats(const float* __restrict__ A,
                                                  const float* __restrict__ mask,
                                                  float* rowsum, float* rownorm) {
  __shared__ float sh[256];
  int b = blockIdx.x / cKU, k = blockIdx.x % cKU, t = threadIdx.x;
  const float* Ab = gptr(A) + (long long)b * cN * cKU + k;
  const float* mb = gptr(mask) + (long long)b * cN;
  float p1 = 0.f, p2 = 0.f;
  for (int n = t; n < cN; n += 256) {
    float a = Ab[(long long)n * cKU] * mb[n];
    p1 += a; p2 += a * a;
  }
  float s1 = bsum(p1, sh);
  float s2 = bsum(p2, sh);
  if (t == 0) { rowsum[b * cKU + k] = s1; rownorm[b * cKU + k] = sqrtf(s2); }
}
__global__ __launch_bounds__(256) void k_occ2(const float* rowsum, float* accum) {
  __shared__ float sh[256];
  int b = blockIdx.x, t = threadIdx.x;
  float p = 0.f;
  for (int k = t; k < cKU; k += 256) p += rowsum[b * cKU + k];
  float tot = fmaxf(bsum(p, sh), 1e-8f);
  float q = 0.f;
  for (int k = t; k < cKU; k += 256) {
    float v = rowsum[b * cKU + k] / tot - 1.0f / (float)cKU;
    q += v * v;
  }
  float sq = bsum(q, sh);
  if (t == 0) atomicAdd(accum, sq);
}
__global__ __launch_bounds__(256) void k_col(const float* __restrict__ Gm,
                                             const float* __restrict__ rownorm,
                                             float* accum) {
  __shared__ float sh[256];
  long long tot = (long long)cB * cKU * cKU;
  float p = 0.f;
  for (long long i = (long long)blockIdx.x * blockDim.x + threadIdx.x; i < tot;
       i += (long long)gridDim.x * blockDim.x) {
    int b = (int)(i / ((long long)cKU * cKU));
    int rem = (int)(i % ((long long)cKU * cKU));
    int k = rem / cKU, l = rem % cKU;
    if (k != l) {
      float v = Gm[i] / (fmaxf(rownorm[b * cKU + k], 1e-8f) *
                         fmaxf(rownorm[b * cKU + l], 1e-8f));
      p += v * v;
    }
  }
  float s = bsum(p, sh);
  if (threadIdx.x == 0) atomicAdd(accum, s);
}
__global__ void k_finalize(const float* acc, float* out) {
  float occ_mse = acc[0] / (float)(cB * cKU);
  float col = acc[1] / ((float)cB * cKU * cKU);
  out[OFF_OMSE] = occ_mse;
  out[OFF_COL]  = col;
  out[OFF_TOT]  = cW_KL * out[OFF_KL] + cW_ENT * out[OFF_ENT] +
                  cW_OCC * occ_mse + cW_COL * col;
}

// ---------------- host-side GEMM helpers ----------------
static inline void gemm_fast(hipStream_t st, const float* A, const float* B,
                             const float* bias, float* C, int M, int Nn, int K,
                             long long sAm, long long sBn, long long sCm,
                             long long bsA, long long bsB, long long bsC, int nb) {
  dim3 g(((M + 63) / 64) * ((Nn + 15) / 16), nb);
  wmma_gemm_fast<<<g, dim3(32), 0, st>>>(A, B, bias, C, M, Nn, K,
                                         sAm, sBn, sCm, bsA, bsB, bsC);
}
static inline void gemm_strided(hipStream_t st, const float* A, const float* B,
                                float* C, int M, int Nn, int K,
                                long long sAm, long long sAk, long long sBn, long long sBk,
                                long long sCm, long long sCn,
                                long long bsA, long long bsB, long long bsC, int nb) {
  dim3 g(((M + 15) / 16) * ((Nn + 15) / 16), nb);
  wmma_gemm_strided<<<g, dim3(32), 0, st>>>(A, B, C, M, Nn, K, sAm, sAk,
                                            sBn, sBk, sCm, sCn, bsA, bsB, bsC);
}

extern "C" void kernel_launch(void* const* d_in, const int* in_sizes, int n_in,
                              void* d_out, int out_size, void* d_ws, size_t ws_size,
                              hipStream_t stream) {
  (void)in_sizes; (void)n_in; (void)out_size; (void)ws_size;
  const float* s    = (const float*)d_in[0];
  const float* mu   = (const float*)d_in[1];
  const float* Sg   = (const float*)d_in[2];
  const float* mask = (const float*)d_in[3];
  const float* pool = (const float*)d_in[4];
  const float* gpool= (const float*)d_in[5];
  const float* gW   = (const float*)d_in[6];
  const float* qW   = (const float*)d_in[7];
  const float* kW   = (const float*)d_in[8];
  const float* wih  = (const float*)d_in[9];
  const float* whh  = (const float*)d_in[10];
  const float* bih  = (const float*)d_in[11];
  const float* bhh  = (const float*)d_in[12];
  const float* lng  = (const float*)d_in[13];
  const float* lnb  = (const float*)d_in[14];
  const float* w1   = (const float*)d_in[15];
  const float* b1   = (const float*)d_in[16];
  const float* w2   = (const float*)d_in[17];
  const float* b2   = (const float*)d_in[18];
  const float* wgeo = (const float*)d_in[19];
  const float* gsr  = (const float*)d_in[20];
  const float* gbias= (const float*)d_in[21];

  float* out = (float*)d_out;
  float* A_out = out + OFF_A;
  float* slots = out + OFF_SLOTS;
  float* mu_k  = out + OFF_MUK;
  float* sig_k = out + OFF_SIGK;
  float* idxF  = out + OFF_IDX;

  float* w = (float*)d_ws;
  long long o = 0;
  auto alloc = [&](long long nfl) { float* p = w + o; o += nfl; return p; };
  float* s_global = alloc((long long)cB * cC);
  float* meanw    = alloc(cB * 3);
  float* stdw     = alloc(cB * 3);
  float* sigavg   = alloc(cB * 9);
  float* scores   = alloc((long long)cB * cKM);
  float* geo_off  = alloc((long long)cB * cKU * 3);
  float* occ      = alloc((long long)cB * cKU);
  int*   idxI     = (int*)alloc((long long)cB * cKU);
  float* k_sem    = alloc((long long)cB * cN * cC);
  float* q_sem    = alloc((long long)cB * cKU * cC);
  float* lsem     = alloc((long long)cB * cN * cKU);
  float* slot_in  = alloc((long long)cB * cKU * cC);
  float* gi       = alloc((long long)cB * cKU * 3 * cC);
  float* gh       = alloc((long long)cB * cKU * 3 * cC);
  float* xm       = alloc((long long)cB * cKU * cC);
  float* h1       = alloc((long long)cB * cKU * 4 * cC);
  float* mo       = alloc((long long)cB * cKU * cC);
  float* rowsum   = alloc(cB * cKU);
  float* rownorm  = alloc(cB * cKU);
  float* Gm       = alloc((long long)cB * cKU * cKU);
  float* hist     = alloc(cKM);
  float* acc      = alloc(8);

  // re-zero accumulators every call (graph replays do not re-poison)
  k_zero<<<8, 256, 0, stream>>>(hist, (long long)cKM + 8);

  k_stats<<<cB, 256, 0, stream>>>(s, mu, Sg, mask, s_global, meanw, stdw, sigavg);
  k_gating<<<cB, 256, 0, stream>>>(s_global, gW, pool, scores);
  k_topk<<<cB, 256, 0, stream>>>(scores, pool, gpool, idxI, idxF, slots, geo_off);
  k_init<<<cB, 256, 0, stream>>>(meanw, stdw, sigavg, geo_off, mu_k, sig_k);

  // k_sem = s @ proj_k_W^T  (M=B*N, N=C, K=C)
  gemm_fast(stream, s, kW, nullptr, k_sem, cB * cN, cC, cC,
            cC, cC, cC, 0, 0, 0, 1);

  for (int it = 0; it < 3; ++it) {
    // q_sem = slots @ proj_q_W^T  (M=B*KU)
    gemm_fast(stream, slots, qW, nullptr, q_sem, cB * cKU, cC, cC,
              cC, cC, cC, 0, 0, 0, 1);
    // logits_sem[b] = k_sem[b] @ q_sem[b]^T  (batched, M=N, N=KU, K=C)
    gemm_fast(stream, k_sem, q_sem, nullptr, lsem, cN, cKU, cC,
              cC, cC, cKU,
              (long long)cN * cC, (long long)cKU * cC, (long long)cN * cKU, cB);
    k_geo_softmax<<<cB * cN, 32, 0, stream>>>(lsem, mu, Sg, mask, mu_k, sig_k,
                                              gsr, gbias, wgeo, A_out);
    k_update<<<cB * cKU, 256, 0, stream>>>(A_out, mu, Sg, mu_k, sig_k, occ);
    // slot_in[b] = A[b]^T @ s[b]  (M=KU, N=C, K=N)  [strided-K path]
    gemm_strided(stream, A_out, s, slot_in, cKU, cC, cN,
                 1, cKU, 1, cC, cC, 1,
                 (long long)cN * cKU, (long long)cN * cC, (long long)cKU * cC, cB);
    k_divocc<<<684, 256, 0, stream>>>(slot_in, occ);
    // GRU gates
    gemm_fast(stream, slot_in, wih, bih, gi, cB * cKU, 3 * cC, cC,
              cC, cC, 3 * cC, 0, 0, 0, 1);
    gemm_fast(stream, slots, whh, bhh, gh, cB * cKU, 3 * cC, cC,
              cC, cC, 3 * cC, 0, 0, 0, 1);
    k_gru<<<684, 256, 0, stream>>>(gi, gh, slots);
    k_ln<<<cB * cKU, 256, 0, stream>>>(slots, lng, lnb, xm);
    // MLP
    gemm_fast(stream, xm, w1, b1, h1, cB * cKU, 4 * cC, cC,
              cC, cC, 4 * cC, 0, 0, 0, 1);
    k_gelu<<<2736, 256, 0, stream>>>(h1);
    gemm_fast(stream, h1, w2, b2, mo, cB * cKU, cC, 4 * cC,
              4 * cC, 4 * cC, cC, 0, 0, 0, 1);
    k_add<<<684, 256, 0, stream>>>(slots, mo);
  }

  // losses
  k_hist<<<(cB * cKU + 255) / 256, 256, 0, stream>>>(idxI, hist);
  k_usage<<<1, 256, 0, stream>>>(hist, out);
  k_colstats<<<cB * cKU, 256, 0, stream>>>(A_out, mask, rowsum, rownorm);
  k_occ2<<<cB, 256, 0, stream>>>(rowsum, &acc[0]);
  // Gm[b] = X[b] @ X[b]^T with X rows = A columns (M=N=KU, K=N)  [strided-K path]
  gemm_strided(stream, A_out, A_out, Gm, cKU, cKU, cN,
               1, cKU, 1, cKU, cKU, 1,
               (long long)cN * cKU, (long long)cN * cKU, (long long)cKU * cKU, cB);
  k_col<<<256, 256, 0, stream>>>(Gm, rownorm, &acc[1]);
  k_finalize<<<1, 1, 0, stream>>>(acc, out);
}
